// CausalSelfAttention_30365418782934
// MI455X (gfx1250) — matrix operand order repaired and checked by reading, not compiled, not run on hardware
//
#include <hip/hip_runtime.h>

typedef __attribute__((ext_vector_type(16))) __bf16 v16bf;
typedef __attribute__((ext_vector_type(8)))  __bf16 v8bf;
typedef __attribute__((ext_vector_type(8)))  float  v8f;

// round-to-nearest-even f32 -> bf16 (bit trick)
__device__ __forceinline__ unsigned short f2bf(float f) {
  unsigned int u = __float_as_uint(f);
  u += 0x7FFFu + ((u >> 16) & 1u);
  return (unsigned short)(u >> 16);
}

// build a v16bf fragment from two 16-byte chunks
__device__ __forceinline__ v16bf ld_frag(const unsigned short* p0, const unsigned short* p1) {
  v8bf lo = *(const v8bf*)p0;
  v8bf hi = *(const v8bf*)p1;
  v16bf r;
#pragma unroll
  for (int i = 0; i < 8; ++i) { r[i] = lo[i]; r[i + 8] = hi[i]; }
  return r;
}

__device__ __forceinline__ v8f wmma_bf16(v16bf a, v16bf b, v8f c) {
  // (neg_a, A, neg_b, B, c_mod, C, reuse_a, reuse_b)
  return __builtin_amdgcn_wmma_f32_16x16x32_bf16(false, a, false, b, (short)0, c, false, false);
}

// ---------------------------------------------------------------------------
// Bulk f32 -> bf16 conversion (4 elements/thread, vectorized)
// ---------------------------------------------------------------------------
__global__ __launch_bounds__(256)
void cvt_f32_bf16(const float* __restrict__ src, unsigned short* __restrict__ dst, int n4) {
  const int i = blockIdx.x * 256 + threadIdx.x;
  if (i >= n4) return;
  const float4 f = ((const float4*)src)[i];
  uint2 p;
  p.x = (unsigned)f2bf(f.x) | ((unsigned)f2bf(f.y) << 16);
  p.y = (unsigned)f2bf(f.z) | ((unsigned)f2bf(f.w) << 16);
  ((uint2*)dst)[i] = p;
}

// ---------------------------------------------------------------------------
// GEMM: C[M,N](f32) = A[M,K](bf16) x B[K,N](bf16) + bias[N](f32).
// Tile 128x128 per block (256 thr = 8 waves; wave tile 32x64 = 2x4 WMMA tiles).
// A tile staged global->LDS with CDNA5 async copies (ASYNCcnt); B tile staged
// transposed (element-granularity) through registers.
// ---------------------------------------------------------------------------
__global__ __launch_bounds__(256)
void gemm_bias_bf16wmma(const unsigned short* __restrict__ A,
                        const unsigned short* __restrict__ B,
                        const float* __restrict__ bias, float* __restrict__ C,
                        int M, int N, int K) {
  __shared__ __align__(16) unsigned short As[128 * 40];  // stride 40 -> conflict-free frag reads
  __shared__ __align__(16) unsigned short Bs[128 * 40];  // transposed: Bs[n][k]

  const int tid  = threadIdx.x;
  const int lane = tid & 31;
  const int wave = tid >> 5;
  const int wm = wave & 3;          // 4 waves along M
  const int wn = wave >> 2;         // 2 waves along N
  const int ln = lane & 15;
  const int hs = (lane >> 4) & 1;   // half-wave select
  const int bm = blockIdx.y * 128;
  const int bn = blockIdx.x * 128;

  const v8f vzero = {0.f, 0.f, 0.f, 0.f, 0.f, 0.f, 0.f, 0.f};
  v8f acc[2][4];
#pragma unroll
  for (int a = 0; a < 2; ++a)
#pragma unroll
    for (int i = 0; i < 4; ++i) acc[a][i] = vzero;

  const int arow = tid >> 1;            // 0..127
  const int acol = (tid & 1) * 16;      // 0 or 16 (bf16 elems; 32B per thread)
  const int brow = tid >> 3;            // 0..31
  const int bcol = (tid & 7) * 16;      // 0..112

  for (int k0 = 0; k0 < K; k0 += 32) {
    // ---- stage A tile (128 x 32 bf16): async global->LDS, 2 x 16B per thread
    {
      const unsigned short* g = A + (size_t)(bm + arow) * K + k0 + acol;
      const unsigned long long g64 = (unsigned long long)(uintptr_t)g;
      const unsigned l0 = (unsigned)(uintptr_t)&As[arow * 40 + acol];
      asm volatile("global_load_async_to_lds_b128 %0, %1, off\n\t"
                   "global_load_async_to_lds_b128 %2, %3, off"
                   :: "v"(l0), "v"(g64), "v"(l0 + 16u), "v"(g64 + 16ull)
                   : "memory");
    }
    // ---- stage B tile (32 x 128 bf16) transposed into Bs[n][k]
    {
      const uint4* g = (const uint4*)(B + (size_t)(k0 + brow) * N + bn + bcol);
      const uint4 w0 = g[0], w1 = g[1];
      const unsigned ww[8] = {w0.x, w0.y, w0.z, w0.w, w1.x, w1.y, w1.z, w1.w};
#pragma unroll
      for (int i = 0; i < 8; ++i) {
        Bs[(bcol + 2 * i + 0) * 40 + brow] = (unsigned short)(ww[i] & 0xFFFFu);
        Bs[(bcol + 2 * i + 1) * 40 + brow] = (unsigned short)(ww[i] >> 16);
      }
    }
    asm volatile("s_wait_asynccnt 0x0" ::: "memory");  // this wave's async copies landed
    __syncthreads();                                   // all waves' staging visible

    // ---- fragments (A and B both in contiguous k order)
    v16bf af[2], bfv[4];
#pragma unroll
    for (int a = 0; a < 2; ++a) {
      const unsigned short* p = &As[(wm * 32 + a * 16 + ln) * 40];
      af[a] = ld_frag(p + hs * 16, p + hs * 16 + 8);      // K=e+hs*16 (contiguous)
    }
#pragma unroll
    for (int i = 0; i < 4; ++i) {
      const unsigned short* p = &Bs[(wn * 64 + i * 16 + ln) * 40 + hs * 16];
      bfv[i] = ld_frag(p, p + 8);                          // K=e+hs*16 (contiguous)
    }
#pragma unroll
    for (int a = 0; a < 2; ++a)
#pragma unroll
      for (int i = 0; i < 4; ++i)
        acc[a][i] = wmma_bf16(af[a], bfv[i], acc[a][i]);
    __syncthreads();
  }

  // ---- epilogue: +bias, store f32 (C-layout: col=ln, rows r + hs*8)
#pragma unroll
  for (int i = 0; i < 4; ++i) {
    const int col = bn + wn * 64 + i * 16 + ln;
    const float bv = bias[col];
#pragma unroll
    for (int a = 0; a < 2; ++a) {
      const int row0 = bm + wm * 32 + a * 16 + hs * 8;
#pragma unroll
      for (int r = 0; r < 8; ++r)
        C[(size_t)(row0 + r) * N + col] = acc[a][i][r] + bv;
    }
  }
}

// ---------------------------------------------------------------------------
// Post-QKV: value mix, RMS-norm(q,k), rotary (position = head index!),
// emit Q,K bf16 [b][h][t][d], V transposed bf16 [b][h][d][t], v1 passthrough.
// One block (128 thr) per (b,t,h); thread = d.
// ---------------------------------------------------------------------------
__global__ __launch_bounds__(128)
void post_qkv(const float* __restrict__ qkv, const float* __restrict__ v1,
              const float* __restrict__ lamb_p,
              unsigned short* __restrict__ Qh, unsigned short* __restrict__ Kh,
              unsigned short* __restrict__ Vt, float* __restrict__ v1_out) {
  const int Tn = 1024, Cn = 2048, H = 16, D = 128;
  const int bid = blockIdx.x;
  const int h = bid & 15;
  const int t = (bid >> 4) & 1023;
  const int b = bid >> 14;
  const int d = threadIdx.x;

  __shared__ float sq[128], sk[128], wred[8];

  const size_t row = ((size_t)b * Tn + t) * (3 * Cn);
  const float q = qkv[row + h * D + d];
  const float k = qkv[row + Cn + h * D + d];
  const float v = qkv[row + 2 * Cn + h * D + d];
  const size_t vi = (((size_t)b * Tn + t) * H + h) * D + d;
  const float v1v = v1[vi];
  const float lam = lamb_p[0];
  const float val = (1.0f - lam) * v + lam * v1v;
  v1_out[vi] = v1v;  // second tuple output

  float s2q = q * q, s2k = k * k;
#pragma unroll
  for (int o = 16; o > 0; o >>= 1) {
    s2q += __shfl_xor(s2q, o, 32);
    s2k += __shfl_xor(s2k, o, 32);
  }
  const int wv = threadIdx.x >> 5;
  if ((threadIdx.x & 31) == 0) { wred[wv] = s2q; wred[4 + wv] = s2k; }
  __syncthreads();
  const float sumq = wred[0] + wred[1] + wred[2] + wred[3];
  const float sumk = wred[4] + wred[5] + wred[6] + wred[7];
  const float rq = rsqrtf(sumq * (1.0f / 128.0f) + 1e-6f);
  const float rk = rsqrtf(sumk * (1.0f / 128.0f) + 1e-6f);
  sq[d] = q * rq;
  sk[d] = k * rk;
  __syncthreads();

  // rotary: reference builds cos/sin over n_pos = H and indexes by head
  const int j = d & 63;
  const float invf = exp2f(-((float)(2 * j) * (1.0f / 128.0f)) * 13.287712379549449f);
  const float ang = (float)h * invf;
  const float cs = __cosf(ang), sn = __sinf(ang);
  float qr, kr;
  if (d < 64) { qr =  sq[d] * cs + sq[d + 64] * sn;  kr =  sk[d] * cs + sk[d + 64] * sn; }
  else        { qr = -sq[d - 64] * sn + sq[d] * cs;  kr = -sk[d - 64] * sn + sk[d] * cs; }

  const size_t bh = (size_t)b * H + h;
  Qh[(bh * Tn + t) * D + d] = f2bf(qr);
  Kh[(bh * Tn + t) * D + d] = f2bf(kr);
  Vt[(bh * D + d) * Tn + t] = f2bf(val);
}

// ---------------------------------------------------------------------------
// Flash attention: block = 4 waves = 64 q rows per (b,h); wave owns 16 q rows.
// S = Q K^T via WMMA (k-chunks of 32), online softmax, P V via WMMA.
// Output written directly as bf16 (A operand of the proj GEMM).
// ---------------------------------------------------------------------------
__global__ __launch_bounds__(128)
void attn_fwd(const unsigned short* __restrict__ Qh, const unsigned short* __restrict__ Kh,
              const unsigned short* __restrict__ Vt, unsigned short* __restrict__ attnB) {
  const int Tn = 1024, D = 128;
  const int bid = blockIdx.x;
  const int qt = bid & 15;     // q tile of 64
  const int bh = bid >> 4;     // b*16+h
  const int lane = threadIdx.x & 31;
  const int wave = threadIdx.x >> 5;
  const int ln = lane & 15;
  const int hs = (lane >> 4) & 1;
  const int qb = qt * 64 + wave * 16;

  __shared__ __align__(16) unsigned short P[4][16 * 32];  // per-wave P staging

  // Q fragments: 4 chunks over D, held in registers for the whole k loop
  const unsigned short* Qrow = Qh + ((size_t)bh * Tn + qb + ln) * D;
  v16bf qf[4];
#pragma unroll
  for (int c = 0; c < 4; ++c) {
    const unsigned short* p0 = Qrow + c * 32 + hs * 16;
    qf[c] = ld_frag(p0, p0 + 8);
  }

  const v8f vzero = {0.f, 0.f, 0.f, 0.f, 0.f, 0.f, 0.f, 0.f};
  v8f o[8];
#pragma unroll
  for (int n = 0; n < 8; ++n) o[n] = vzero;
  float mrow[8], lrow[8];
#pragma unroll
  for (int r = 0; r < 8; ++r) { mrow[r] = -3.0e38f; lrow[r] = 0.0f; }

  const float scale = 0.08838834764831845f;  // 1/sqrt(128)

  for (int k0 = 0; k0 <= qb + 15; k0 += 32) {
    // ---- S tiles: cols [k0,k0+16) and [k0+16,k0+32)
    v8f Sa = vzero, Sb = vzero;
#pragma unroll
    for (int c = 0; c < 4; ++c) {
      const unsigned short* Ka = Kh + ((size_t)bh * Tn + k0 + ln) * D + c * 32 + hs * 16;
      Sa = wmma_bf16(qf[c], ld_frag(Ka, Ka + 8), Sa);
      const unsigned short* Kb = Kh + ((size_t)bh * Tn + k0 + 16 + ln) * D + c * 32 + hs * 16;
      Sb = wmma_bf16(qf[c], ld_frag(Kb, Kb + 8), Sb);
    }

    // ---- causal mask + online softmax (row stats live in 16-lane halves)
#pragma unroll
    for (int r = 0; r < 8; ++r) {
      const int qrow = qb + hs * 8 + r;
      float sa = Sa[r] * scale; if (k0 + ln > qrow)       sa = -3.0e38f;
      float sb = Sb[r] * scale; if (k0 + 16 + ln > qrow)  sb = -3.0e38f;
      float mx = fmaxf(sa, sb);
#pragma unroll
      for (int mk = 8; mk >= 1; mk >>= 1) mx = fmaxf(mx, __shfl_xor(mx, mk, 16));
      const float mnew = fmaxf(mrow[r], mx);
      const float corr = __expf(mrow[r] - mnew);
      const float pa = __expf(sa - mnew);
      const float pb = __expf(sb - mnew);
      float rs = pa + pb;
#pragma unroll
      for (int mk = 8; mk >= 1; mk >>= 1) rs += __shfl_xor(rs, mk, 16);
      lrow[r] = lrow[r] * corr + rs;
      mrow[r] = mnew;
#pragma unroll
      for (int n = 0; n < 8; ++n) o[n][r] *= corr;
      const int prow = hs * 8 + r;
      P[wave][prow * 32 + ln]      = f2bf(pa);
      P[wave][prow * 32 + 16 + ln] = f2bf(pb);
    }
    asm volatile("s_wait_dscnt 0x0" ::: "memory");  // wave-local LDS RAW before frag read

    // ---- P (16x32) as A fragment, V chunks as B fragments (Vt is [d][t])
    const unsigned short* pr = &P[wave][ln * 32];
    const v16bf pf = ld_frag(pr + hs * 16, pr + hs * 16 + 8);
#pragma unroll
    for (int n = 0; n < 8; ++n) {
      const unsigned short* Vp = Vt + ((size_t)bh * D + n * 16 + ln) * Tn + k0 + hs * 16;
      o[n] = wmma_bf16(pf, ld_frag(Vp, Vp + 8), o[n]);
    }
  }

  // ---- normalize, store attn as bf16 [b][t][h*128+d] (proj GEMM A operand)
  const int h = bh & 15, b = bh >> 4;
#pragma unroll
  for (int r = 0; r < 8; ++r) {
    const float inv = 1.0f / lrow[r];
    const int trow = qb + hs * 8 + r;
    const size_t orow = ((size_t)b * Tn + trow) * 2048 + h * 128;
#pragma unroll
    for (int n = 0; n < 8; ++n)
      attnB[orow + n * 16 + ln] = f2bf(o[n][r] * inv);
  }
}

// ---------------------------------------------------------------------------
extern "C" void kernel_launch(void* const* d_in, const int* in_sizes, int n_in,
                              void* d_out, int out_size, void* d_ws, size_t ws_size,
                              hipStream_t stream) {
  (void)in_sizes; (void)n_in; (void)out_size; (void)ws_size;
  const float* x      = (const float*)d_in[0];
  const float* v1     = (const float*)d_in[1];
  const float* W_qkv  = (const float*)d_in[2];
  const float* b_qkv  = (const float*)d_in[3];
  const float* W_proj = (const float*)d_in[4];
  const float* b_proj = (const float*)d_in[5];
  const float* lamb   = (const float*)d_in[6];

  const int Bn = 4, Tn = 1024, Cn = 2048, H = 16;
  const int M = Bn * Tn;        // 4096
  const int Nqkv = 3 * Cn;      // 6144

  // workspace layout (bytes):
  // qkv f32 | Qh bf16 | Kh bf16 | Vt bf16 | attn bf16 | x bf16 | Wqkv bf16 | Wproj bf16
  char* ws = (char*)d_ws;
  float*          qkv   = (float*)(ws);                        // 100,663,296 B
  unsigned short* Qh    = (unsigned short*)(ws + 100663296);   //  16,777,216 B
  unsigned short* Kh    = (unsigned short*)(ws + 117440512);   //  16,777,216 B
  unsigned short* Vt    = (unsigned short*)(ws + 134217728);   //  16,777,216 B
  unsigned short* attnB = (unsigned short*)(ws + 150994944);   //  16,777,216 B
  unsigned short* xbf   = (unsigned short*)(ws + 167772160);   //  16,777,216 B
  unsigned short* wqbf  = (unsigned short*)(ws + 184549376);   //  25,165,824 B
  unsigned short* wpbf  = (unsigned short*)(ws + 209715200);   //   8,388,608 B

  float* out   = (float*)d_out;
  float* v1out = out + (size_t)M * Cn;

  // one-time f32 -> bf16 operand conversion (halves GEMM operand traffic)
  cvt_f32_bf16<<<(M * Cn / 4 + 255) / 256, 256, 0, stream>>>(x, xbf, M * Cn / 4);
  cvt_f32_bf16<<<(Cn * Nqkv / 4 + 255) / 256, 256, 0, stream>>>(W_qkv, wqbf, Cn * Nqkv / 4);
  cvt_f32_bf16<<<(Cn * Cn / 4 + 255) / 256, 256, 0, stream>>>(W_proj, wpbf, Cn * Cn / 4);

  gemm_bias_bf16wmma<<<dim3(Nqkv / 128, M / 128), 256, 0, stream>>>(
      xbf, wqbf, b_qkv, qkv, M, Nqkv, Cn);
  post_qkv<<<Bn * Tn * H, 128, 0, stream>>>(qkv, v1, lamb, Qh, Kh, Vt, v1out);
  attn_fwd<<<Bn * H * (Tn / 64), 128, 0, stream>>>(Qh, Kh, Vt, attnB);
  gemm_bias_bf16wmma<<<dim3(Cn / 128, M / 128), 256, 0, stream>>>(
      attnB, wpbf, b_proj, out, M, Cn, Cn);
}
